// GraphModule_14972255994536
// MI455X (gfx1250) — matrix operand order repaired and checked by reading, not compiled, not run on hardware
//
#include <hip/hip_runtime.h>
#include <hip/hip_bf16.h>

// ---------------- problem constants (from reference) ----------------
#define NN      50000
#define FF      256
#define H1n     4
#define D1n     32
#define HD1n    128          // H1*D1
#define D2n     64
#define EEn     1000000
#define NUM_NEW 10000

// ---------------- WMMA types ----------------
typedef __attribute__((ext_vector_type(16))) __bf16 v16bf;
typedef __attribute__((ext_vector_type(8)))  __bf16 v8bf;
typedef __attribute__((ext_vector_type(8)))  float  v8f;

// =====================================================================
// GEMM: C[M,Nout] = A[M,K] (bf16, row-major) x B[K,Nout] where B is
// supplied transposed as Bt[Nout,K] (bf16, row-major). f32 accumulate.
// One wave per 16x16 C tile; K-loop of v_wmma_f32_16x16x32_bf16.
// Requires M%16==0, K%32==0, Nout%16==0 (true here: 50000,256/128,128/64).
// =====================================================================
template <int K, int NOUT>
__global__ void wmma_gemm_bf16(const __bf16* __restrict__ A,
                               const __bf16* __restrict__ Bt,
                               float* __restrict__ C, int M)
{
    int wave = (int)((blockIdx.x * blockDim.x + threadIdx.x) >> 5);
    int lane = threadIdx.x & 31;
    constexpr int ntN = NOUT >> 4;
    int ntiles = (M >> 4) * ntN;
    if (wave >= ntiles) return;
    int m0 = (wave / ntN) << 4;
    int n0 = (wave % ntN) << 4;

    // A fragment (16-bit A 16x32 ISA layout):
    //   lanes 0-15 : row M=m0+lane,    K offsets {0..7, 16..23}
    //   lanes16-31 : row M=m0+lane-16, K offsets {8..15, 24..31}
    int arow  = m0 + (lane & 15);
    int ahalf = (lane >> 4) ? 8 : 0;
    // B fragment: lane holds column N=n0+(lane&15);
    //   lanes 0-15 hold K {0..7, 16..23}, lanes 16-31 hold K {8..15, 24..31}
    int bcol  = n0 + (lane & 15);
    int bhalf = (lane >> 4) ? 8 : 0;

    const __bf16* Arow = A  + (size_t)arow * K;
    const __bf16* Brow = Bt + (size_t)bcol * K;

    v8f acc = {};
#pragma unroll
    for (int k0 = 0; k0 < K; k0 += 32) {
        union { v16bf v; v8bf h[2]; } a;
        a.h[0] = *(const v8bf*)(Arow + k0 + ahalf);        // K base..base+7
        a.h[1] = *(const v8bf*)(Arow + k0 + ahalf + 16);   // K base+16..base+23
        union { v16bf v; v8bf h[2]; } bu;
        bu.h[0] = *(const v8bf*)(Brow + k0 + bhalf);
        bu.h[1] = *(const v8bf*)(Brow + k0 + bhalf + 16);
        v16bf b = bu.v;
        acc = __builtin_amdgcn_wmma_f32_16x16x32_bf16(
                  false, a.v, false, b, (short)0, acc, false, false);
    }

    // C/D layout: VGPR r -> row m0 + r + 8*(lane>=16), col n0 + (lane&15)
    int mbase = m0 + ((lane >> 4) << 3);
    float* Cp = C + (size_t)mbase * NOUT + n0 + (lane & 15);
#pragma unroll
    for (int r = 0; r < 8; ++r) Cp[(size_t)r * NOUT] = acc[r];
}

// ---------------- helpers ----------------
__global__ void f32_to_bf16_kernel(const float* __restrict__ in,
                                   __bf16* __restrict__ out, int n)
{
    int t = blockIdx.x * blockDim.x + threadIdx.x;
    if (t < n) out[t] = (__bf16)in[t];
}

// W[rows,cols] (f32) -> Wt[cols,rows] (bf16)
template <int COLS>
__global__ void transpose_to_bf16_kernel(const float* __restrict__ W,
                                         __bf16* __restrict__ Wt, int rows)
{
    int t = blockIdx.x * blockDim.x + threadIdx.x;
    if (t < rows * COLS) {
        int r = t / COLS, c = t % COLS;
        Wt[c * rows + r] = (__bf16)W[t];
    }
}

// alpha_s[n,h] = <hx[n,h,:], a_src[h,:]>, alpha_d likewise
template <int HH, int DD>
__global__ void att_scores_kernel(const float* __restrict__ hx,
                                  const float* __restrict__ a_src,
                                  const float* __restrict__ a_dst,
                                  float* __restrict__ as_, float* __restrict__ ad_,
                                  int n)
{
    int t = blockIdx.x * blockDim.x + threadIdx.x;
    if (t >= n * HH) return;
    int node = t / HH, h = t % HH;
    const float* v = hx + (size_t)node * (HH * DD) + h * DD;
    float s = 0.f, d = 0.f;
#pragma unroll
    for (int j = 0; j < DD; ++j) {
        float x = v[j];
        s += x * a_src[h * DD + j];
        d += x * a_dst[h * DD + j];
    }
    as_[t] = s; ad_[t] = d;
}

// order-preserving float -> uint key (for atomic segment-max)
__device__ __forceinline__ unsigned fkey(float f) {
    unsigned b = __float_as_uint(f);
    return (b & 0x80000000u) ? ~b : (b | 0x80000000u);
}
__device__ __forceinline__ float funkey(unsigned k) {
    unsigned b = (k & 0x80000000u) ? (k & 0x7FFFFFFFu) : ~k;
    return __uint_as_float(b);
}

// pass 1: logits e = leaky_relu(as[src]+ad[dst], 0.2); segment max (atomic key-max)
template <int HH>
__global__ void edge_logits_max_kernel(const int* __restrict__ src,
                                       const int* __restrict__ dst,
                                       const float* __restrict__ as_,
                                       const float* __restrict__ ad_,
                                       float* __restrict__ elog,
                                       unsigned* __restrict__ emaxk)
{
    int t = blockIdx.x * blockDim.x + threadIdx.x;
    if (t >= EEn * HH) return;
    int e = t / HH, h = t % HH;          // HH is power of 2 -> shifts
    int s = src[e], d = dst[e];
    float v = as_[s * HH + h] + ad_[d * HH + h];
    v = v > 0.f ? v : 0.2f * v;
    elog[t] = v;
    atomicMax(&emaxk[d * HH + h], fkey(v));
}

// pass 2: elog <- exp(elog - max[dst]); denom[dst] += elog
template <int HH>
__global__ void edge_exp_sum_kernel(const int* __restrict__ dst,
                                    float* __restrict__ elog,
                                    const unsigned* __restrict__ emaxk,
                                    float* __restrict__ denom)
{
    int t = blockIdx.x * blockDim.x + threadIdx.x;
    if (t >= EEn * HH) return;
    int e = t / HH, h = t % HH;
    int d = dst[e];
    float m = funkey(emaxk[d * HH + h]);
    float a = __expf(elog[t] - m);
    elog[t] = a;
    atomicAdd(&denom[d * HH + h], a);
}

// pass 3: out[dst] += hx[src] * (elog / (denom[dst]+1e-16)); float4 per thread.
// 32 consecutive lanes cover one 512B hx row -> fully coalesced gather,
// contiguous-row atomic scatter (L2 float atomics).
template <int HH, int DD>
__global__ void aggregate_kernel(const int* __restrict__ src,
                                 const int* __restrict__ dst,
                                 const float* __restrict__ hx,
                                 const float* __restrict__ aexp,
                                 const float* __restrict__ denom,
                                 float* __restrict__ out)
{
    constexpr int HD = HH * DD;
    constexpr int Q  = HD >> 2;          // float4 chunks per edge
    long t = (long)blockIdx.x * blockDim.x + threadIdx.x;
    if (t >= (long)EEn * Q) return;
    int e  = (int)(t / Q);               // Q is power of 2 -> shift
    int c4 = ((int)t & (Q - 1)) << 2;
    int h  = c4 / DD;
    int s = src[e], d = dst[e];
    float w = aexp[e * HH + h] / (denom[d * HH + h] + 1e-16f);
    float4 v = *(const float4*)(hx + (size_t)s * HD + c4);
    float* o = out + (size_t)d * HD + c4;
    atomicAdd(o + 0, v.x * w);
    atomicAdd(o + 1, v.y * w);
    atomicAdd(o + 2, v.z * w);
    atomicAdd(o + 3, v.w * w);
}

// x1b = bf16(elu(out1 + b1)) — feeds the layer-2 WMMA GEMM
template <int C>
__global__ void elu_bias_bf16_kernel(const float* __restrict__ out1,
                                     const float* __restrict__ b,
                                     __bf16* __restrict__ x1b, int n)
{
    int t = blockIdx.x * blockDim.x + threadIdx.x;
    if (t >= n) return;
    float v = out1[t] + b[t & (C - 1)];
    v = v > 0.f ? v : (__expf(v) - 1.f);
    x1b[t] = (__bf16)v;
}

// d_out = out2[last NUM_NEW rows] + b2   (mean over H=1 head is identity)
template <int C>
__global__ void final_copy_kernel(const float* __restrict__ out2,
                                  const float* __restrict__ b2,
                                  float* __restrict__ dout,
                                  int nrows, int rowoff)
{
    int t = blockIdx.x * blockDim.x + threadIdx.x;
    if (t >= nrows * C) return;
    int r = t / C, c = t & (C - 1);
    dout[t] = out2[(size_t)(rowoff + r) * C + c] + b2[c];
}

// ---------------- host orchestration ----------------
static inline size_t align256(size_t x) { return (x + 255) & ~(size_t)255; }
static inline int nblk(long n, int b) { return (int)((n + b - 1) / b); }

extern "C" void kernel_launch(void* const* d_in, const int* in_sizes, int n_in,
                              void* d_out, int out_size, void* d_ws, size_t ws_size,
                              hipStream_t stream)
{
    (void)in_sizes; (void)n_in; (void)out_size; (void)ws_size;

    const float* x    = (const float*)d_in[0];
    const int*   src  = (const int*)  d_in[1];
    const int*   dst  = (const int*)  d_in[2];
    const float* W1   = (const float*)d_in[3];
    const float* as1  = (const float*)d_in[4];
    const float* ad1  = (const float*)d_in[5];
    const float* b1   = (const float*)d_in[6];
    const float* W2   = (const float*)d_in[7];
    const float* as2  = (const float*)d_in[8];
    const float* ad2  = (const float*)d_in[9];
    const float* b2   = (const float*)d_in[10];
    float*       out  = (float*)d_out;

    // ---- workspace carve-up (~140 MB; fits MI455X's 192 MB L2) ----
    char* ws = (char*)d_ws;
    size_t cur = 0;
    auto take = [&](size_t bytes) { size_t o = cur; cur = align256(cur + bytes); return o; };

    __bf16*   xb    = (__bf16*)  (ws + take((size_t)NN * FF    * 2));
    __bf16*   w1t   = (__bf16*)  (ws + take((size_t)HD1n * FF  * 2));   // [128,256]
    __bf16*   w2t   = (__bf16*)  (ws + take((size_t)D2n * HD1n * 2));   // [64,128]
    float*    hx1   = (float*)   (ws + take((size_t)NN * HD1n  * 4));
    float*    sc_s1 = (float*)   (ws + take((size_t)NN * H1n   * 4));
    float*    sc_d1 = (float*)   (ws + take((size_t)NN * H1n   * 4));
    unsigned* emax1 = (unsigned*)(ws + take((size_t)NN * H1n   * 4));
    float*    den1  = (float*)   (ws + take((size_t)NN * H1n   * 4));
    float*    ae1   = (float*)   (ws + take((size_t)EEn * H1n  * 4));
    float*    out1  = (float*)   (ws + take((size_t)NN * HD1n  * 4));
    __bf16*   x1b   = (__bf16*)  (ws + take((size_t)NN * HD1n  * 2));
    float*    hx2   = (float*)   (ws + take((size_t)NN * D2n   * 4));
    float*    sc_s2 = (float*)   (ws + take((size_t)NN * 4));
    float*    sc_d2 = (float*)   (ws + take((size_t)NN * 4));
    unsigned* emax2 = (unsigned*)(ws + take((size_t)NN * 4));
    float*    den2  = (float*)   (ws + take((size_t)NN * 4));
    float*    ae2   = (float*)   (ws + take((size_t)EEn * 4));
    float*    out2  = (float*)   (ws + take((size_t)NN * D2n   * 4));

    const int B = 256;

    // accumulator buffers must be zero at every call (graph replay safe)
    hipMemsetAsync(out1,  0, (size_t)NN * HD1n * 4, stream);
    hipMemsetAsync(den1,  0, (size_t)NN * H1n  * 4, stream);
    hipMemsetAsync(emax1, 0, (size_t)NN * H1n  * 4, stream);  // key(-inf) > 0
    hipMemsetAsync(out2,  0, (size_t)NN * D2n  * 4, stream);
    hipMemsetAsync(den2,  0, (size_t)NN * 4, stream);
    hipMemsetAsync(emax2, 0, (size_t)NN * 4, stream);

    // ---- precision prep ----
    f32_to_bf16_kernel<<<nblk((long)NN * FF, B), B, 0, stream>>>(x, xb, NN * FF);
    transpose_to_bf16_kernel<HD1n><<<nblk(FF * HD1n, B), B, 0, stream>>>(W1, w1t, FF);
    transpose_to_bf16_kernel<D2n><<<nblk(HD1n * D2n, B), B, 0, stream>>>(W2, w2t, HD1n);

    // ================= layer 1 =================
    {
        int tiles = (NN / 16) * (HD1n / 16);          // 25000 waves
        wmma_gemm_bf16<FF, HD1n><<<nblk((long)tiles * 32, B), B, 0, stream>>>(xb, w1t, hx1, NN);

        att_scores_kernel<H1n, D1n><<<nblk((long)NN * H1n, B), B, 0, stream>>>(hx1, as1, ad1, sc_s1, sc_d1, NN);

        edge_logits_max_kernel<H1n><<<nblk((long)EEn * H1n, B), B, 0, stream>>>(src, dst, sc_s1, sc_d1, ae1, emax1);
        edge_exp_sum_kernel<H1n><<<nblk((long)EEn * H1n, B), B, 0, stream>>>(dst, ae1, emax1, den1);

        aggregate_kernel<H1n, D1n><<<nblk((long)EEn * (HD1n / 4), B), B, 0, stream>>>(src, dst, hx1, ae1, den1, out1);

        elu_bias_bf16_kernel<HD1n><<<nblk((long)NN * HD1n, B), B, 0, stream>>>(out1, b1, x1b, NN * HD1n);
    }

    // ================= layer 2 (H=1, D=64) =================
    {
        int tiles = (NN / 16) * (D2n / 16);           // 12500 waves
        wmma_gemm_bf16<HD1n, D2n><<<nblk((long)tiles * 32, B), B, 0, stream>>>(x1b, w2t, hx2, NN);

        att_scores_kernel<1, D2n><<<nblk((long)NN, B), B, 0, stream>>>(hx2, as2, ad2, sc_s2, sc_d2, NN);

        edge_logits_max_kernel<1><<<nblk((long)EEn, B), B, 0, stream>>>(src, dst, sc_s2, sc_d2, ae2, emax2);
        edge_exp_sum_kernel<1><<<nblk((long)EEn, B), B, 0, stream>>>(dst, ae2, emax2, den2);

        aggregate_kernel<1, D2n><<<nblk((long)EEn * (D2n / 4), B), B, 0, stream>>>(src, dst, hx2, ae2, den2, out2);
    }

    final_copy_kernel<D2n><<<nblk((long)NUM_NEW * D2n, B), B, 0, stream>>>(out2, b2, out, NUM_NEW, NN - NUM_NEW);
}
